// DiscreteDecisionEngine_87385404605279
// MI455X (gfx1250) — matrix operand order repaired and checked by reading, not compiled, not run on hardware
//
#include <hip/hip_runtime.h>
#include <hip/hip_bf16.h>

// ---------------------------------------------------------------------------
// logits = x @ (W @ G)^T + b, where G = blockdiag of 4x4 Hamilton matrices
// built from tanh(geodesic_weights). Fold W' = W @ G once (f16), then do one
// WMMA GEMM [M,1024] x [1024,256] with fp32->f16 convert-on-load for x and
// f32 accumulation. HBM-bound: ~320 MB moved -> ~14 us floor on MI455X.
// ---------------------------------------------------------------------------

#define STATE_DIM   1024
#define NUM_ACTIONS 256

typedef _Float16 v16h __attribute__((ext_vector_type(16)));
typedef float    v8f  __attribute__((ext_vector_type(8)));
typedef float    v4f  __attribute__((ext_vector_type(4)));

// ---------------------------------------------------------------------------
// Prep: Wp[a][4n+d] = sum_c W[a][4n+c] * M(q_n)[c][d], stored f16 row-major.
// M(q) = [[ w,-i,-j,-k],
//         [ i, w,-k, j],
//         [ j, k, w,-i],
//         [ k,-j, i, w]]
// One thread per (action, quaternion) pair: 65536 threads total.
// ---------------------------------------------------------------------------
__global__ __launch_bounds__(256) void fold_weights_kernel(
    const float* __restrict__ gw,      // [256*4] geodesic weights
    const float* __restrict__ W,       // [256, 1024]
    _Float16*    __restrict__ Wp)      // [256, 1024] f16 out
{
    const int t = blockIdx.x * blockDim.x + threadIdx.x;   // 0..65535
    const int a = t >> 8;                                  // action
    const int n = t & 255;                                 // quaternion slot

    const float w = tanhf(gw[4 * n + 0]);
    const float i = tanhf(gw[4 * n + 1]);
    const float j = tanhf(gw[4 * n + 2]);
    const float k = tanhf(gw[4 * n + 3]);

    const float* Wr = W + (size_t)a * STATE_DIM + 4 * n;
    const float W0 = Wr[0], W1 = Wr[1], W2 = Wr[2], W3 = Wr[3];

    _Float16* o = Wp + (size_t)a * STATE_DIM + 4 * n;
    o[0] = (_Float16)( w * W0 + i * W1 + j * W2 + k * W3);
    o[1] = (_Float16)(-i * W0 + w * W1 + k * W2 - j * W3);
    o[2] = (_Float16)(-j * W0 - k * W1 + w * W2 + i * W3);
    o[3] = (_Float16)(-k * W0 + j * W1 - i * W2 + w * W3);
}

// ---------------------------------------------------------------------------
// GEMM: out[M,256] = X[M,1024](fp32, cvt->f16) * Wp^T (f16) + bias, f32 accum.
// Block = 256 threads = 8 waves arranged 2(M) x 4(N); block tile 32 x 256.
// Wave tile 16 x 64 -> 4 x v_wmma_f32_16x16x32_f16 accumulators.
// All 4 B fragments are loaded into distinct registers before the WMMA chain
// so the loads clause together and the waits resolve incrementally.
// ---------------------------------------------------------------------------
__global__ __launch_bounds__(256) void ddq_gemm_wmma_kernel(
    const float*    __restrict__ X,     // [M, 1024]
    const _Float16* __restrict__ Wp,    // [256, 1024]
    const float*    __restrict__ bias,  // [256]
    float*          __restrict__ out,   // [M, 256]
    int M)
{
    const int lane   = threadIdx.x & 31;
    const int wid    = threadIdx.x >> 5;   // 0..7
    const int wave_m = wid & 1;            // 0..1  -> 16-row sub-tile
    const int wave_n = wid >> 1;           // 0..3  -> 64-col sub-tile

    const int half   = lane >> 4;          // 0 or 1 (lane group)
    const int l16    = lane & 15;

    // A fragment addressing (16x32 f16 layout): row M = l16,
    // K chunk [half*16, +16) (matches B).
    const int mrow  = blockIdx.x * 32 + wave_m * 16 + l16;
    const int akoff = half * 16;
    const float* xrow = X + (size_t)mrow * STATE_DIM;

    // B fragment addressing (32x16 f16 layout): col N = l16,
    // K chunk [half*16, +16) -> one contiguous v16h load per tile.
    // Tiles t=0..3 sit at constant offsets t*16*STATE_DIM from bbase.
    const _Float16* bbase =
        Wp + (size_t)(wave_n * 64 + l16) * STATE_DIM + half * 16;

    v8f acc[4] = {};   // 4 accumulators of 16x16 f32 (8 VGPRs each)

    for (int k0 = 0; k0 < STATE_DIM; k0 += 32) {
        // Speculative prefetch 2 iterations ahead (global_prefetch_b8);
        // invalid tail addresses are silently dropped by hardware.
        __builtin_prefetch(xrow + k0 + 64, 0, 3);

        // ---- A fragment: fp32 load + convert to f16 in-register ----
        v4f a0 = *(const v4f*)(xrow + k0 + akoff + 0);
        v4f a1 = *(const v4f*)(xrow + k0 + akoff + 4);
        v4f a2 = *(const v4f*)(xrow + k0 + akoff + 8);
        v4f a3 = *(const v4f*)(xrow + k0 + akoff + 12);
        v16h afrag;
#pragma unroll
        for (int e = 0; e < 4; ++e) {
            afrag[e]      = (_Float16)a0[e];
            afrag[4 + e]  = (_Float16)a1[e];
            afrag[8 + e]  = (_Float16)a2[e];
            afrag[12 + e] = (_Float16)a3[e];
        }

        // ---- Load all 4 B fragments first (L2-resident, 512 KB total) ----
        v16h bfrag[4];
#pragma unroll
        for (int t = 0; t < 4; ++t) {
            bfrag[t] = *(const v16h*)(bbase + (size_t)t * 16 * STATE_DIM + k0);
        }

        // ---- WMMA chain: waits resolve incrementally per fragment ----
#pragma unroll
        for (int t = 0; t < 4; ++t) {
            acc[t] = __builtin_amdgcn_wmma_f32_16x16x32_f16(
                /*neg_a=*/false, afrag,
                /*neg_b=*/false, bfrag[t],
                /*c_mod=*/(short)0, acc[t],
                /*reuse_a=*/false, /*reuse_b=*/false);
        }
    }

    // ---- Store: C/D layout -> M = r + 8*half, N = l16 (+ bias) ----
    const int mbase = blockIdx.x * 32 + wave_m * 16 + half * 8;
#pragma unroll
    for (int t = 0; t < 4; ++t) {
        const int ncol = wave_n * 64 + t * 16 + l16;
        const float bv = bias[ncol];
        float* optr = out + (size_t)mbase * NUM_ACTIONS + ncol;
#pragma unroll
        for (int r = 0; r < 8; ++r) {
            optr[(size_t)r * NUM_ACTIONS] = acc[t][r] + bv;
        }
    }
}

// ---------------------------------------------------------------------------
// Host launcher. Inputs: d_in[0]=x [B,1024] f32, d_in[1]=geodesic [1,256,4],
// d_in[2]=W [256,1024], d_in[3]=b [256]. Output: logits [B,256] f32.
// ---------------------------------------------------------------------------
extern "C" void kernel_launch(void* const* d_in, const int* in_sizes, int n_in,
                              void* d_out, int out_size, void* d_ws, size_t ws_size,
                              hipStream_t stream)
{
    const float* x    = (const float*)d_in[0];
    const float* gw   = (const float*)d_in[1];
    const float* W    = (const float*)d_in[2];
    const float* bias = (const float*)d_in[3];
    float* out        = (float*)d_out;

    const int M = in_sizes[0] / STATE_DIM;   // 65536

    // Folded weights live in workspace: 256*1024 f16 = 512 KB.
    _Float16* Wp = (_Float16*)d_ws;

    // 1) Fold tanh + Hamilton into W (tiny: 65536 threads).
    fold_weights_kernel<<<NUM_ACTIONS, 256, 0, stream>>>(gw, W, Wp);

    // 2) WMMA GEMM over the batch: block tile 32 rows x 256 cols.
    ddq_gemm_wmma_kernel<<<M / 32, 256, 0, stream>>>(x, Wp, bias, out, M);
}
